// transformer_80272938762533
// MI455X (gfx1250) — matrix operand order repaired and checked by reading, not compiled, not run on hardware
//
#include <hip/hip_runtime.h>
#include <hip/hip_bf16.h>

// MI455X / gfx1250 implementation: wave32, WMMA f32_16x16x32_f16 for all
// channel contractions, f32 accumulation, f32 stats/normalization.

typedef __attribute__((ext_vector_type(16))) _Float16 v16h;
typedef __attribute__((ext_vector_type(8)))  _Float16 v8h;
typedef __attribute__((ext_vector_type(8)))  float    v8f;
typedef __attribute__((ext_vector_type(4)))  float    v4f;

#define B_    16
#define N_    2000
#define K9    9
#define CI    256
#define CO    128
#define NK    (N_*K9)        // 18000 spatial elements per (b, ci)
#define LOCS  (B_*N_)        // 32000 (b,n) locations

// ---- WMMA fragment loaders (contiguous k = hi*16 + s order, same for A and B) ------------------
// A (16x32 f16, row-major source, row = lane%16, hi = lane/16):
//   elements 0..7  <- K = hi*16 + 0..7
//   elements 8..15 <- K = hi*16 + 8 + 0..7
__device__ __forceinline__ v16h load_a_frag(const _Float16* __restrict__ wrow,
                                            int koff, int hi) {
    union { v16h v; v8h h[2]; } u;
    u.h[0] = *(const v8h*)(wrow + koff + hi * 16);
    u.h[1] = *(const v8h*)(wrow + koff + hi * 16 + 8);
    return u.v;
}
// B (32x16 f16, column-major source: col = lane%16 contiguous over K):
//   elements 0..15 <- K = hi*16 + 0..15  (one 32-byte contiguous chunk)
__device__ __forceinline__ v16h load_b_frag(const _Float16* __restrict__ xcol,
                                            int koff, int hi) {
    return *(const v16h*)(xcol + koff + hi * 16);
}

// ---- Stage 1: per-(b,ci) sum / sumsq of x_local ----------------------------
__global__ __launch_bounds__(256) void stats_xlocal_kernel(
    const float* __restrict__ x, float* __restrict__ sums) {
    __shared__ float red[512];
    size_t base = (size_t)blockIdx.x * NK;
    float s = 0.f, q = 0.f;
    for (int j = threadIdx.x; j < NK; j += 256) {
        float v = x[base + j]; s += v; q += v * v;
    }
    red[threadIdx.x] = s; red[256 + threadIdx.x] = q;
    __syncthreads();
    for (int st = 128; st > 0; st >>= 1) {
        if ((int)threadIdx.x < st) {
            red[threadIdx.x]       += red[threadIdx.x + st];
            red[256 + threadIdx.x] += red[256 + threadIdx.x + st];
        }
        __syncthreads();
    }
    if (threadIdx.x == 0) {
        sums[blockIdx.x * 2]     = red[0];
        sums[blockIdx.x * 2 + 1] = red[256];
    }
}

// inorm(1e-3) followed by bnorm(1e-5) collapses to y = s*x + t per (b,c):
// batch mean of inorm output is exactly 0; batch var = mean_b(v/(v+1e-3)).
__global__ void finalize_xlocal_kernel(const float* __restrict__ sums,
                                       float* __restrict__ s1,
                                       float* __restrict__ t1) {
    int i = threadIdx.x;            // channel 0..255
    float varc = 0.f;
    for (int b = 0; b < B_; ++b) {
        float m = sums[(b * CI + i) * 2] * (1.f / NK);
        float v = sums[(b * CI + i) * 2 + 1] * (1.f / NK) - m * m;
        varc += v / (v + 1e-3f);
    }
    varc *= (1.f / B_);
    float rc = rsqrtf(varc + 1e-5f);
    for (int b = 0; b < B_; ++b) {
        float m = sums[(b * CI + i) * 2] * (1.f / NK);
        float v = sums[(b * CI + i) * 2 + 1] * (1.f / NK) - m * m;
        float s = rsqrtf(v + 1e-3f) * rc;
        s1[b * CI + i] = s;
        t1[b * CI + i] = -m * s;
    }
}

// ---- Weight prep: Wk' = wk@w_att1, Wv' = wv@w_att1 (fuses xl away) --------
__global__ __launch_bounds__(256) void prep_comb_kernel(
    const float* __restrict__ wk, const float* __restrict__ bk,
    const float* __restrict__ wv, const float* __restrict__ bv,
    const float* __restrict__ w_att1, const float* __restrict__ b_att1,
    _Float16* __restrict__ Wk16, _Float16* __restrict__ Wv16,
    float* __restrict__ bkc, float* __restrict__ bvc) {
    int idx = blockIdx.x * 256 + threadIdx.x;   // 0..65535
    int which = idx >> 15;
    int c = (idx >> 8) & 127;
    int i = idx & 255;
    const float* wm = which ? wv : wk;
    float acc = 0.f;
    for (int j = 0; j < CO; ++j) acc += wm[c * CO + j] * w_att1[j * CI + i];
    (which ? Wv16 : Wk16)[c * CI + i] = (_Float16)acc;
    if (i == 0) {
        float bb = (which ? bv : bk)[c];
        for (int j = 0; j < CO; ++j) bb += wm[c * CO + j] * b_att1[j];
        (which ? bvc : bkc)[c] = bb;
    }
}

// Convert small weights to f16; stack [w_l1; w_right] so u1 and the x1
// shortcut come from one M=256 GEMM.
__global__ void prep_conv_kernel(
    const float* __restrict__ wq, const float* __restrict__ wl1,
    const float* __restrict__ wr, const float* __restrict__ wl2,
    const float* __restrict__ wl3, const float* __restrict__ bl1,
    const float* __restrict__ br,
    _Float16* __restrict__ wq16, _Float16* __restrict__ wstack,
    float* __restrict__ bstack, _Float16* __restrict__ wl2_16,
    _Float16* __restrict__ wl3_16) {
    int idx = blockIdx.x * 256 + threadIdx.x;
    if (idx < 16384) { wq16[idx] = (_Float16)wq[idx]; return; }
    idx -= 16384;
    if (idx < 16384) { wl2_16[idx] = (_Float16)wl2[idx]; return; }
    idx -= 16384;
    if (idx < 16384) { wl3_16[idx] = (_Float16)wl3[idx]; return; }
    idx -= 16384;
    if (idx < 65536) {
        int c = idx >> 8, i = idx & 255;
        wstack[idx] = (_Float16)((c < 128) ? wl1[c * CI + i]
                                           : wr[(c - 128) * CI + i]);
        return;
    }
    idx -= 65536;
    if (idx < 256) bstack[idx] = (idx < 128) ? bl1[idx] : br[idx - 128];
}

// x_row [b][c][n] -> h [b][n][c] f16 (lower 128 channels), via LDS transpose
__global__ __launch_bounds__(256) void xrow_to_h_kernel(
    const float* __restrict__ xr, _Float16* __restrict__ h) {
    __shared__ float tile[16][17];
    int blk = blockIdx.x;                 // b*8*125 + cb*125 + nb
    int nb = blk % 125; blk /= 125;
    int cb = blk % 8;   int b = blk / 8;
    int tx = threadIdx.x & 15, ty = threadIdx.x >> 4;
    tile[ty][tx] = xr[((size_t)(b * CO + cb * 16 + ty)) * N_ + nb * 16 + tx];
    __syncthreads();
    int n = nb * 16 + ty, c = cb * 16 + tx;
    h[((size_t)(b * N_ + n)) * CI + c] = (_Float16)tile[tx][ty];
}

// ---- Generic WMMA GEMM: Y[loc][c] = sum_i W[c][i] * X[loc][i] + bias[c] ----
// W: f16 [M][kc] row-major; X: f16 [loc][ldx] (channels contiguous);
// Y: f32 [loc][ldy]. Block = 8 waves; wave w handles M-tile blockIdx.y*8+w,
// column tile = blockIdx.x (16 locations). No LDS needed: A chunks and the
// 32-byte B chunks are contiguous in global; W stays hot in the 192MB L2.
__global__ __launch_bounds__(256) void gemm16_kernel(
    const _Float16* __restrict__ W, const float* __restrict__ bias,
    const _Float16* __restrict__ X, int ldx, int kc,
    float* __restrict__ Y, int ldy) {
    const int lane = threadIdx.x & 31;
    const int wave = threadIdx.x >> 5;
    const int mt = blockIdx.y * 8 + wave;
    const int colbase = blockIdx.x * 16;
    const int lm = lane & 15, hi = lane >> 4;
    const _Float16* wrow = W + (size_t)(mt * 16 + lm) * kc;
    const _Float16* xcol = X + (size_t)(colbase + lm) * ldx;
    v8f acc = {};
    for (int ks = 0; ks < kc; ks += 32) {
        v16h a = load_a_frag(wrow, ks, hi);
        v16h b = load_b_frag(xcol, ks, hi);
        acc = __builtin_amdgcn_wmma_f32_16x16x32_f16(
            false, a, false, b, (short)0, acc, false, false);
    }
    const int c0 = mt * 16 + 8 * hi;
    float* yp = Y + (size_t)(colbase + lm) * ldy + c0;
#pragma unroll
    for (int r = 0; r < 8; ++r) yp[r] = acc[r] + bias[c0 + r];
}

// ---- Fused attention: normalize+relu x_local tile -> dual k/v WMMA GEMM
// (K=256) -> softmax over K9 -> out_local written as h upper half (f16) -----
__global__ __launch_bounds__(256) void attn_kernel(
    const float* __restrict__ xloc, const float* __restrict__ s1,
    const float* __restrict__ t1,
    const _Float16* __restrict__ Wk16, const _Float16* __restrict__ Wv16,
    const float* __restrict__ bkc, const float* __restrict__ bvc,
    const float* __restrict__ q, _Float16* __restrict__ h) {
    // 72KB LDS, reused: phase1 Bt[col][256] f16; phase3 kv[(m*128+c)][144] f16
    __shared__ __align__(32) _Float16 lds[36864];
    const int b  = blockIdx.x / 125;
    const int nb = blockIdx.x % 125;

    // phase 1: stage r = relu(s*x + t) as f16, column-major (col = nl*9+kk)
    {
        int i = threadIdx.x;
        float s = s1[b * CI + i], t = t1[b * CI + i];
        const float* src = xloc + (size_t)(b * CI + i) * NK + nb * 144;
        for (int col = 0; col < 144; col += 4) {
            v4f x = *(const v4f*)(src + col);
#pragma unroll
            for (int u = 0; u < 4; ++u) {
                float r = fmaf(x[u], s, t);
                lds[(col + u) * 256 + i] = (_Float16)(r > 0.f ? r : 0.f);
            }
        }
    }
    __syncthreads();

    // phase 2: k = Wk'*r, v = Wv'*r   (wave w -> channels w*16..w*16+15,
    // 9 column tiles x 2 matrices = 18 v8f accumulators)
    const int lane = threadIdx.x & 31;
    const int wave = threadIdx.x >> 5;
    const int lm = lane & 15, hi = lane >> 4;
    v8f acck[9] = {}, accv[9] = {};
    const _Float16* wkrow = Wk16 + (size_t)(wave * 16 + lm) * CI;
    const _Float16* wvrow = Wv16 + (size_t)(wave * 16 + lm) * CI;
    for (int ks = 0; ks < CI; ks += 32) {
        v16h ak = load_a_frag(wkrow, ks, hi);
        v16h av = load_a_frag(wvrow, ks, hi);
#pragma unroll
        for (int ct = 0; ct < 9; ++ct) {
            v16h bf = load_b_frag(&lds[(ct * 16 + lm) * 256], ks, hi);
            acck[ct] = __builtin_amdgcn_wmma_f32_16x16x32_f16(
                false, ak, false, bf, (short)0, acck[ct], false, false);
            accv[ct] = __builtin_amdgcn_wmma_f32_16x16x32_f16(
                false, av, false, bf, (short)0, accv[ct], false, false);
        }
    }
    __syncthreads();   // everyone done reading Bt before overwriting with kv

    // phase 3: accumulators (+bias) -> LDS  kv[k/v][c][col] f16
    {
        int cbase = wave * 16 + 8 * hi;
#pragma unroll
        for (int ct = 0; ct < 9; ++ct) {
            int col = ct * 16 + lm;
#pragma unroll
            for (int r = 0; r < 8; ++r) {
                int c = cbase + r;
                lds[(size_t)c * 144 + col]         = (_Float16)(acck[ct][r] + bkc[c]);
                lds[(size_t)(128 + c) * 144 + col] = (_Float16)(accv[ct][r] + bvc[c]);
            }
        }
    }
    __syncthreads();

    // phase 4: softmax over K9 of q*k, weighted sum of v -> h[.., 128+c]
    for (int p = 0; p < 8; ++p) {
        int pair = p * 256 + threadIdx.x;     // 0..2047 = (nl,c)
        int c = pair & 127, nl = pair >> 7;
        int n = nb * 16 + nl;
        float qv = q[((size_t)(b * N_ + n)) * CO + c];
        float sc[9], mx = -1e30f;
#pragma unroll
        for (int kk = 0; kk < 9; ++kk) {
            float kx = (float)lds[(size_t)c * 144 + nl * 9 + kk];
            sc[kk] = qv * kx;
            mx = fmaxf(mx, sc[kk]);
        }
        float den = 0.f, num = 0.f;
#pragma unroll
        for (int kk = 0; kk < 9; ++kk) {
            float e = __expf(sc[kk] - mx);
            den += e;
            num += e * (float)lds[(size_t)(128 + c) * 144 + nl * 9 + kk];
        }
        h[((size_t)(b * N_ + n)) * CI + 128 + c] = (_Float16)(num / den);
    }
}

// ---- per-(b,c) stats over N for the ResNet norms ---------------------------
__global__ __launch_bounds__(256) void stats_act_kernel(
    const float* __restrict__ u, int ldu, float* __restrict__ mv) {
    __shared__ float red[512];
    int b = blockIdx.x;
    int c = threadIdx.x & 127, half = threadIdx.x >> 7;
    float s = 0.f, q = 0.f;
    for (int n = half; n < N_; n += 2) {
        float v = u[((size_t)(b * N_ + n)) * ldu + c];
        s += v; q += v * v;
    }
    red[threadIdx.x] = s; red[256 + threadIdx.x] = q;
    __syncthreads();
    if (half == 0) {
        float st = red[threadIdx.x] + red[threadIdx.x + 128];
        float qt = red[256 + threadIdx.x] + red[256 + threadIdx.x + 128];
        float m = st * (1.f / N_);
        mv[(b * CO + c) * 2]     = m;
        mv[(b * CO + c) * 2 + 1] = qt * (1.f / N_) - m * m;
    }
}

__global__ void finalize_act_kernel(const float* __restrict__ mv,
                                    float* __restrict__ s2,
                                    float* __restrict__ t2) {
    int c = threadIdx.x;                 // 0..127
    float varc = 0.f;
    for (int b = 0; b < B_; ++b) {
        float v = mv[(b * CO + c) * 2 + 1];
        varc += v / (v + 1e-5f);
    }
    varc *= (1.f / B_);
    float rc = rsqrtf(varc + 1e-5f);
    for (int b = 0; b < B_; ++b) {
        float m = mv[(b * CO + c) * 2], v = mv[(b * CO + c) * 2 + 1];
        float s = rsqrtf(v + 1e-5f) * rc;
        s2[b * CO + c] = s;
        t2[b * CO + c] = -m * s;
    }
}

// o = relu_after_add ? relu(norm(u)+x1) : relu(norm(u)) + x1 [+ add]
__global__ __launch_bounds__(256) void apply_norm_kernel(
    const float* __restrict__ u, int ldu,
    const float* __restrict__ s2, const float* __restrict__ t2,
    const float* __restrict__ x1, int ldx1,
    const float* __restrict__ add, int relu_after_add,
    float* __restrict__ o32, _Float16* __restrict__ o16) {
    size_t idx = (size_t)blockIdx.x * 256 + threadIdx.x;   // LOCS*128
    int c = idx & 127;
    size_t loc = idx >> 7;
    int b = (int)(loc / N_);
    float y  = fmaf(u[loc * ldu + c], s2[b * CO + c], t2[b * CO + c]);
    float xv = x1[loc * ldx1 + c];
    float o;
    if (relu_after_add) { o = y + xv; o = o > 0.f ? o : 0.f; }
    else {
        o = (y > 0.f ? y : 0.f) + xv;
        if (add) o += add[loc * CO + c];
    }
    o32[loc * CO + c] = o;
    if (o16) o16[loc * CO + c] = (_Float16)o;
}

// out = x_row + gamma*o3;  final = (out + mean_c(out)) * 0.5
__global__ __launch_bounds__(256) void final_kernel(
    const float* __restrict__ xr, const float* __restrict__ o3,
    const float* __restrict__ gamma, float* __restrict__ out) {
    __shared__ float red[256];
    int n2 = threadIdx.x >> 7, c = threadIdx.x & 127;
    size_t loc = (size_t)blockIdx.x * 2 + n2;
    int b = (int)(loc / N_), n = (int)(loc % N_);
    float val = xr[((size_t)(b * CO + c)) * N_ + n] + gamma[0] * o3[loc * CO + c];
    red[threadIdx.x] = val;
    __syncthreads();
    for (int st = 64; st > 0; st >>= 1) {
        if (c < st) red[threadIdx.x] += red[threadIdx.x + st];
        __syncthreads();
    }
    float mean = red[n2 * 128] * (1.f / CO);
    out[((size_t)(b * CO + c)) * N_ + n] = (val + mean) * 0.5f;
}

extern "C" void kernel_launch(void* const* d_in, const int* in_sizes, int n_in,
                              void* d_out, int out_size, void* d_ws, size_t ws_size,
                              hipStream_t stream) {
    (void)in_sizes; (void)n_in; (void)out_size; (void)ws_size;
    const float* x_row   = (const float*)d_in[0];
    const float* x_local = (const float*)d_in[1];
    const float* w_att1  = (const float*)d_in[2];
    const float* b_att1  = (const float*)d_in[3];
    const float* wq      = (const float*)d_in[4];
    const float* bq      = (const float*)d_in[5];
    const float* wk      = (const float*)d_in[6];
    const float* bk      = (const float*)d_in[7];
    const float* wv      = (const float*)d_in[8];
    const float* bv      = (const float*)d_in[9];
    const float* w_right = (const float*)d_in[10];
    const float* b_right = (const float*)d_in[11];
    const float* w_l1    = (const float*)d_in[12];
    const float* b_l1    = (const float*)d_in[13];
    const float* w_l2    = (const float*)d_in[14];
    const float* b_l2    = (const float*)d_in[15];
    const float* w_l3    = (const float*)d_in[16];
    const float* b_l3    = (const float*)d_in[17];
    const float* gamma1  = (const float*)d_in[18];
    float* out = (float*)d_out;

    // ---- workspace bump allocator (aligned 256B) ----
    char* ws = (char*)d_ws;
    size_t off = 0;
    auto alloc = [&](size_t bytes) -> char* {
        char* p = ws + off;
        off += (bytes + 255) & ~(size_t)255;
        return p;
    };
    float*    statsum = (float*)alloc(4096 * 2 * 4);
    float*    s1      = (float*)alloc(B_ * CI * 4);
    float*    t1      = (float*)alloc(B_ * CI * 4);
    _Float16* Wk16    = (_Float16*)alloc(CO * CI * 2);
    _Float16* Wv16    = (_Float16*)alloc(CO * CI * 2);
    float*    bkc     = (float*)alloc(CO * 4);
    float*    bvc     = (float*)alloc(CO * 4);
    _Float16* wq16    = (_Float16*)alloc(CO * CO * 2);
    _Float16* wstack  = (_Float16*)alloc(256 * CI * 2);
    float*    bstack  = (float*)alloc(256 * 4);
    _Float16* wl2_16  = (_Float16*)alloc(CO * CO * 2);
    _Float16* wl3_16  = (_Float16*)alloc(CO * CO * 2);
    _Float16* h       = (_Float16*)alloc((size_t)LOCS * CI * 2);
    float*    qb      = (float*)alloc((size_t)LOCS * CO * 4);
    float*    ux      = (float*)alloc((size_t)LOCS * 256 * 4);  // [u1 | x1]
    float*    u2      = (float*)alloc((size_t)LOCS * CO * 4);   // also u3
    float*    o1_32   = (float*)alloc((size_t)LOCS * CO * 4);   // also o3
    _Float16* o1_16   = (_Float16*)alloc((size_t)LOCS * CO * 2);
    float*    o2_32   = (float*)alloc((size_t)LOCS * CO * 4);
    _Float16* o2_16   = (_Float16*)alloc((size_t)LOCS * CO * 2);
    float*    mv      = (float*)alloc(B_ * CO * 2 * 4);
    float*    s2      = (float*)alloc(B_ * CO * 4);
    float*    t2      = (float*)alloc(B_ * CO * 4);

    // 1) collapsed inorm+bnorm scales for x_local
    stats_xlocal_kernel<<<4096, 256, 0, stream>>>(x_local, statsum);
    finalize_xlocal_kernel<<<1, 256, 0, stream>>>(statsum, s1, t1);

    // 2) weight prep (combined k/v weights; f16 conversions; stacked l1/right)
    prep_comb_kernel<<<256, 256, 0, stream>>>(wk, bk, wv, bv, w_att1, b_att1,
                                              Wk16, Wv16, bkc, bvc);
    prep_conv_kernel<<<449, 256, 0, stream>>>(wq, w_l1, w_right, w_l2, w_l3,
                                              b_l1, b_right, wq16, wstack,
                                              bstack, wl2_16, wl3_16);
    // 3) x_row -> h lower half (f16, location-major)
    xrow_to_h_kernel<<<B_ * 8 * 125, 256, 0, stream>>>(x_row, h);

    // 4) q = wq @ x_row + bq   (WMMA, K=128)
    gemm16_kernel<<<dim3(LOCS / 16, 1), 256, 0, stream>>>(
        wq16, bq, h, CI, CO, qb, CO);

    // 5) fused normalize -> k/v GEMM -> softmax -> out_local (h upper half)
    attn_kernel<<<B_ * 125, 256, 0, stream>>>(x_local, s1, t1, Wk16, Wv16,
                                              bkc, bvc, qb, h);

    // 6) [u1; x1] = [w_l1; w_right] @ h + [b_l1; b_right]   (WMMA, K=256, M=256)
    gemm16_kernel<<<dim3(LOCS / 16, 2), 256, 0, stream>>>(
        wstack, bstack, h, CI, CI, ux, 256);

    // 7) o1 = relu(norm(u1)) + x1
    stats_act_kernel<<<B_, 256, 0, stream>>>(ux, 256, mv);
    finalize_act_kernel<<<1, 128, 0, stream>>>(mv, s2, t2);
    apply_norm_kernel<<<LOCS * CO / 256, 256, 0, stream>>>(
        ux, 256, s2, t2, ux + 128, 256, nullptr, 0, o1_32, o1_16);

    // 8) o2 = relu(norm(w_l2 @ o1 + b_l2)) + o1 + x1
    gemm16_kernel<<<dim3(LOCS / 16, 1), 256, 0, stream>>>(
        wl2_16, b_l2, o1_16, CO, CO, u2, CO);
    stats_act_kernel<<<B_, 256, 0, stream>>>(u2, CO, mv);
    finalize_act_kernel<<<1, 128, 0, stream>>>(mv, s2, t2);
    apply_norm_kernel<<<LOCS * CO / 256, 256, 0, stream>>>(
        u2, CO, s2, t2, ux + 128, 256, o1_32, 0, o2_32, o2_16);

    // 9) o3 = relu(norm(w_l3 @ o2 + b_l3) + x1)   (u3 reuses u2, o3 reuses o1_32)
    gemm16_kernel<<<dim3(LOCS / 16, 1), 256, 0, stream>>>(
        wl3_16, b_l3, o2_16, CO, CO, u2, CO);
    stats_act_kernel<<<B_, 256, 0, stream>>>(u2, CO, mv);
    finalize_act_kernel<<<1, 128, 0, stream>>>(mv, s2, t2);
    apply_norm_kernel<<<LOCS * CO / 256, 256, 0, stream>>>(
        u2, CO, s2, t2, ux + 128, 256, nullptr, 1, o1_32, nullptr);

    // 10) out = (x_row + gamma*o3 + channel-mean)/2
    final_kernel<<<LOCS / 2, 256, 0, stream>>>(x_row, o1_32, gamma1, out);
}
